// DIFAttentionLayer_25915832664860
// MI455X (gfx1250) — matrix operand order repaired and checked by reading, not compiled, not run on hardware
//
#include <hip/hip_runtime.h>

// ---------------------------------------------------------------------------
// CDNA5 (gfx1250) fused DIF-attention: bf16 WMMA 16x16x32, wave32, flash-style
// online softmax with dual (content/category) statistics.
//
// Pipeline (stream-ordered):
//   0) cvt:  content/category/weights f32 -> bf16 workspace (packed cvt)
//   1) proj: 5 projections, 16x64 output tile per wave (1 A-frag x 4 B-frags)
//   2) attn: dual flash attention, 16-query tile per wave, 32-key chunks
//   3) out:  attn @ Wo^T + bo, 16x64 tile per wave, f32 output
// ---------------------------------------------------------------------------

typedef __bf16 v16bf __attribute__((ext_vector_type(16)));
typedef __bf16 v8bf  __attribute__((ext_vector_type(8)));
typedef __bf16 v4bf  __attribute__((ext_vector_type(4)));
typedef float  v8f   __attribute__((ext_vector_type(8)));

union BF16x16 { v16bf v; v8bf h[2]; };

static __device__ __forceinline__ v8f wmma_bf16(v16bf a, v16bf b, v8f c) {
  // D = A(16x32 bf16) * B(32x16 bf16) + C(16x16 f32)
  return __builtin_amdgcn_wmma_f32_16x16x32_bf16(false, a, false, b, (short)0, c,
                                                 false, false);
}

#define NB  4
#define NT  2048
#define ND  256
#define NH  8
#define NHD 32

static constexpr size_t PROJ_ELEMS = (size_t)NB * NH * NT * NHD;  // 2,097,152
static constexpr size_t WMAT_ELEMS = (size_t)ND * ND;             // 65,536
static constexpr size_t QC_OFF  = 0;
static constexpr size_t KC_OFF  = 1 * PROJ_ELEMS;
static constexpr size_t QK_OFF  = 2 * PROJ_ELEMS;
static constexpr size_t KK_OFF  = 3 * PROJ_ELEMS;
static constexpr size_t VT_OFF  = 4 * PROJ_ELEMS;   // V stored transposed [B,H,HD,T]
static constexpr size_t XC_OFF  = 5 * PROJ_ELEMS;   // content bf16 [B*T, D]
static constexpr size_t XK_OFF  = 6 * PROJ_ELEMS;   // category bf16 [B*T, D]
static constexpr size_t WB_OFF  = 7 * PROJ_ELEMS;   // 6 weight mats bf16
// attn output reuses the content-bf16 region: proj consumes XC before attn writes.
static constexpr size_t ATT_OFF = XC_OFF;
// total workspace: 7*PROJ + 6*WMAT elems = 15,073,280 bf16 ~= 30.1 MB

// ---------------------------------------------------------------------------
// Kernel 0: packed f32 -> bf16 conversion (4 elements / thread).
// ---------------------------------------------------------------------------
__global__ __launch_bounds__(256) void dif_cvt_kernel(
    const float* __restrict__ src, __bf16* __restrict__ dst, int n4)
{
  const int i = blockIdx.x * 256 + threadIdx.x;
  if (i < n4) {
    const float4 x = ((const float4*)src)[i];
    v4bf y;
    y[0] = (__bf16)x.x; y[1] = (__bf16)x.y;
    y[2] = (__bf16)x.z; y[3] = (__bf16)x.w;
    ((v4bf*)dst)[i] = y;
  }
}

// ---------------------------------------------------------------------------
// Kernel 1: five projections Y = X @ W^T, all-bf16 operands.
// One wave = one 16x64 output tile: A-frag reused across 4 B-frags.
// ---------------------------------------------------------------------------
__global__ __launch_bounds__(256) void dif_proj_kernel(__bf16* __restrict__ ws)
{
  const int lane = threadIdx.x & 31;
  const int wave = threadIdx.x >> 5;
  const int w    = blockIdx.x * 8 + wave;      // 5 * 512 * 4 = 10240 waves
  const int proj = w >> 11;                    // / 2048
  const int rem  = w & 2047;
  const int m0   = (rem >> 2) << 4;            // row tile (tokens)
  const int n0   = (rem & 3) << 6;             // col group (4 x 16 features)

  const __bf16* X = ws + ((proj < 3) ? XC_OFF : XK_OFF);
  const __bf16* W = ws + WB_OFF + (size_t)proj * WMAT_ELEMS;

  const int l16  = lane & 15;
  const int half = lane >> 4;
  const int arow = m0 + l16;                   // A: row per lane
  const int akb0 = half ? 16 : 0;              // A frag K split: contiguous, matches B
  const int akb1 = half ? 24 : 8;
  const int bkb  = half ? 16 : 0;              // B frag K split

  v8f acc[4] = {{}, {}, {}, {}};
  #pragma unroll
  for (int kc = 0; kc < ND; kc += 32) {
    BF16x16 a;
    const __bf16* xr = X + (size_t)arow * ND + kc;
    a.h[0] = *(const v8bf*)(xr + akb0);
    a.h[1] = *(const v8bf*)(xr + akb1);
    #pragma unroll
    for (int j = 0; j < 4; ++j) {
      BF16x16 b;
      const __bf16* wr = W + (size_t)(n0 + j * 16 + l16) * ND + kc + bkb;
      b.h[0] = *(const v8bf*)(wr);
      b.h[1] = *(const v8bf*)(wr + 8);
      acc[j] = wmma_bf16(a.v, b.v, acc[j]);
    }
  }

  // Scatter D-layout (N = lane%16, M = v + 8*half) into head-split storage.
  #pragma unroll
  for (int j = 0; j < 4; ++j) {
    const int col = n0 + j * 16 + l16;
    const int h   = col >> 5;
    const int hd  = col & 31;
    #pragma unroll
    for (int v = 0; v < 8; ++v) {
      const int m    = m0 + v + 8 * half;
      const int bidx = m >> 11;                // / T
      const int t    = m & (NT - 1);
      const __bf16 val = (__bf16)acc[j][v];
      if (proj == 2) {
        ws[VT_OFF + (((size_t)(bidx * NH + h)) * NHD + hd) * NT + t] = val;  // V^T
      } else {
        const size_t off = (proj == 0) ? QC_OFF : (proj == 1) ? KC_OFF :
                           (proj == 3) ? QK_OFF : KK_OFF;
        ws[off + (((size_t)(bidx * NH + h)) * NT + t) * NHD + hd] = val;
      }
    }
  }
}

// ---------------------------------------------------------------------------
// Online-softmax update for one score type over a 16q x 32k chunk.
// Scores arrive in C/D layout (lane holds col N=l16, rows v+8*half).
// Emits unnormalized exp probabilities into LDS (bf16, [16][32] row-major).
// ---------------------------------------------------------------------------
static __device__ __forceinline__ void softmax_update(
    const v8f& s0, const v8f& s1, int q0, int kc, int half, int l16,
    float scale, float* mrow, float* lrow, v8f& acc0, v8f& acc1,
    __bf16* __restrict__ ldsBuf)
{
  const int key0 = kc + l16;
  const int key1 = key0 + 16;
  #pragma unroll
  for (int v = 0; v < 8; ++v) {
    const int q = q0 + v + 8 * half;
    float x0 = (key0 <= q) ? s0[v] * scale : -__builtin_inff();   // causal mask
    float x1 = (key1 <= q) ? s1[v] * scale : -__builtin_inff();
    float r  = fmaxf(x0, x1);                   // row max across 16 lanes (half-wave)
    r = fmaxf(r, __shfl_xor(r, 1, 32));
    r = fmaxf(r, __shfl_xor(r, 2, 32));
    r = fmaxf(r, __shfl_xor(r, 4, 32));
    r = fmaxf(r, __shfl_xor(r, 8, 32));
    const float mn   = fmaxf(mrow[v], r);
    const float corr = __expf(mrow[v] - mn);    // first chunk: exp(-inf)=0
    const float p0   = __expf(x0 - mn);
    const float p1   = __expf(x1 - mn);
    float rs = p0 + p1;                         // row sum
    rs += __shfl_xor(rs, 1, 32);
    rs += __shfl_xor(rs, 2, 32);
    rs += __shfl_xor(rs, 4, 32);
    rs += __shfl_xor(rs, 8, 32);
    lrow[v] = lrow[v] * corr + rs;
    mrow[v] = mn;
    acc0[v] *= corr;
    acc1[v] *= corr;
    const int mr = v + 8 * half;
    ldsBuf[mr * 32 + l16]      = (__bf16)p0;
    ldsBuf[mr * 32 + 16 + l16] = (__bf16)p1;
  }
}

// ---------------------------------------------------------------------------
// Kernel 2: dual flash attention. One wave per 16-query tile of one (b,h).
// Per 32-key chunk: 4 QK^T WMMAs, dual online softmax, LDS re-layout
// (C/D-layout -> A-fragment layout), 4 AV WMMAs.
// ---------------------------------------------------------------------------
__global__ __launch_bounds__(256) void dif_attn_kernel(
    const float* __restrict__ alpha_p, __bf16* __restrict__ ws)
{
  __shared__ __attribute__((aligned(16))) __bf16 ldsP[8][2][16 * 32];  // 16 KB

  const int lane = threadIdx.x & 31;
  const int wave = threadIdx.x >> 5;
  const int w    = blockIdx.x * 8 + wave;      // 4 * 8 * 128 = 4096 waves
  const int bh   = w >> 7;                     // b*H + h
  const int q0   = (w & 127) << 4;

  const __bf16* Qc = ws + QC_OFF + (size_t)bh * NT * NHD;
  const __bf16* Kc = ws + KC_OFF + (size_t)bh * NT * NHD;
  const __bf16* Qk = ws + QK_OFF + (size_t)bh * NT * NHD;
  const __bf16* Kk = ws + KK_OFF + (size_t)bh * NT * NHD;
  const __bf16* Vt = ws + VT_OFF + (size_t)bh * NHD * NT;
  __bf16* attn     = ws + ATT_OFF;

  const float alpha = 1.0f / (1.0f + __expf(-alpha_p[0]));
  const float scale = 0.17677669529663687f;    // 1/sqrt(HD=32)

  const int l16  = lane & 15;
  const int half = lane >> 4;
  const int kb0  = half ? 16 : 0;
  const int kb1  = half ? 24 : 8;
  const int dkb  = half ? 16 : 0;

  // Q fragments (K = HD = 32, one A-frag each), loaded once.
  BF16x16 aqc, aqk;
  {
    const size_t qoff = (size_t)(q0 + l16) * NHD;
    aqc.h[0] = *(const v8bf*)(Qc + qoff + kb0);
    aqc.h[1] = *(const v8bf*)(Qc + qoff + kb1);
    aqk.h[0] = *(const v8bf*)(Qk + qoff + kb0);
    aqk.h[1] = *(const v8bf*)(Qk + qoff + kb1);
  }

  v8f acc_c0 = {}, acc_c1 = {}, acc_k0 = {}, acc_k1 = {};
  float m_c[8], l_c[8], m_k[8], l_k[8];
  #pragma unroll
  for (int v = 0; v < 8; ++v) {
    m_c[v] = m_k[v] = -__builtin_inff();
    l_c[v] = l_k[v] = 0.0f;
  }

  const int kend = q0 + 16;                    // causal horizon for this tile
  const v8f vzero = {};

  for (int kc = 0; kc < kend; kc += 32) {
    // K fragments: B[k=d, n=key] = K[key, d] -> contiguous 16 bf16 per lane.
    BF16x16 bc0, bc1, bk0, bk1;
    const size_t k0off = (size_t)(kc + l16) * NHD + dkb;
    const size_t k1off = (size_t)(kc + 16 + l16) * NHD + dkb;
    bc0.h[0] = *(const v8bf*)(Kc + k0off);
    bc0.h[1] = *(const v8bf*)(Kc + k0off + 8);
    bc1.h[0] = *(const v8bf*)(Kc + k1off);
    bc1.h[1] = *(const v8bf*)(Kc + k1off + 8);
    bk0.h[0] = *(const v8bf*)(Kk + k0off);
    bk0.h[1] = *(const v8bf*)(Kk + k0off + 8);
    bk1.h[0] = *(const v8bf*)(Kk + k1off);
    bk1.h[1] = *(const v8bf*)(Kk + k1off + 8);

    v8f s_c0 = wmma_bf16(aqc.v, bc0.v, vzero);
    v8f s_c1 = wmma_bf16(aqc.v, bc1.v, vzero);
    v8f s_k0 = wmma_bf16(aqk.v, bk0.v, vzero);
    v8f s_k1 = wmma_bf16(aqk.v, bk1.v, vzero);

    asm volatile("" ::: "memory");             // keep LDS stores after prior loads
    softmax_update(s_c0, s_c1, q0, kc, half, l16, scale, m_c, l_c,
                   acc_c0, acc_c1, &ldsP[wave][0][0]);
    softmax_update(s_k0, s_k1, q0, kc, half, l16, scale, m_k, l_k,
                   acc_k0, acc_k1, &ldsP[wave][1][0]);

    __builtin_amdgcn_wave_barrier();
    asm volatile("s_wait_dscnt 0" ::: "memory");   // cross-lane LDS exchange, same wave

    // Re-read probabilities in A-fragment layout (16q x 32k).
    BF16x16 apc, apk;
    apc.h[0] = *(const v8bf*)&ldsP[wave][0][l16 * 32 + kb0];
    apc.h[1] = *(const v8bf*)&ldsP[wave][0][l16 * 32 + kb1];
    apk.h[0] = *(const v8bf*)&ldsP[wave][1][l16 * 32 + kb0];
    apk.h[1] = *(const v8bf*)&ldsP[wave][1][l16 * 32 + kb1];

    // V fragments from V^T: B[k=key, n=d] = Vt[d, key] -> contiguous per lane.
    BF16x16 bv0, bv1;
    const int krel = kc + (half ? 16 : 0);
    bv0.h[0] = *(const v8bf*)(Vt + (size_t)l16 * NT + krel);
    bv0.h[1] = *(const v8bf*)(Vt + (size_t)l16 * NT + krel + 8);
    bv1.h[0] = *(const v8bf*)(Vt + (size_t)(16 + l16) * NT + krel);
    bv1.h[1] = *(const v8bf*)(Vt + (size_t)(16 + l16) * NT + krel + 8);

    acc_c0 = wmma_bf16(apc.v, bv0.v, acc_c0);
    acc_c1 = wmma_bf16(apc.v, bv1.v, acc_c1);
    acc_k0 = wmma_bf16(apk.v, bv0.v, acc_k0);
    acc_k1 = wmma_bf16(apk.v, bv1.v, acc_k1);
    __builtin_amdgcn_wave_barrier();
  }

  // Blend: alpha*softmax(Ak)@V + (1-alpha)*softmax(Ac)@V, merge heads.
  const int b_idx = bh >> 3;
  const int h_idx = bh & 7;
  #pragma unroll
  for (int v = 0; v < 8; ++v) {
    const float wc = (1.0f - alpha) / l_c[v];
    const float wk = alpha / l_k[v];
    const int q = q0 + v + 8 * half;
    const size_t row = ((size_t)(b_idx * NT + q)) * ND + h_idx * NHD;
    attn[row + l16]      = (__bf16)(wk * acc_k0[v] + wc * acc_c0[v]);
    attn[row + 16 + l16] = (__bf16)(wk * acc_k1[v] + wc * acc_c1[v]);
  }
}

// ---------------------------------------------------------------------------
// Kernel 3: out = attn @ Wo^T + bo (f32 output), 16x64 tile per wave.
// ---------------------------------------------------------------------------
__global__ __launch_bounds__(256) void dif_out_kernel(
    const __bf16* __restrict__ ws, const float* __restrict__ bo,
    float* __restrict__ out)
{
  const int lane = threadIdx.x & 31;
  const int wave = threadIdx.x >> 5;
  const int w    = blockIdx.x * 8 + wave;      // 512 * 4 = 2048 waves
  const int m0   = (w >> 2) << 4;
  const int n0   = (w & 3) << 6;

  const __bf16* attn = ws + ATT_OFF;
  const __bf16* Wo   = ws + WB_OFF + 5 * WMAT_ELEMS;

  const int l16  = lane & 15;
  const int half = lane >> 4;
  const int kb0  = half ? 16 : 0;
  const int kb1  = half ? 24 : 8;
  const int arow = m0 + l16;
  const int bkb  = half ? 16 : 0;

  v8f acc[4] = {{}, {}, {}, {}};
  #pragma unroll
  for (int kc = 0; kc < ND; kc += 32) {
    BF16x16 a;
    const __bf16* ar = attn + (size_t)arow * ND + kc;
    a.h[0] = *(const v8bf*)(ar + kb0);
    a.h[1] = *(const v8bf*)(ar + kb1);
    #pragma unroll
    for (int j = 0; j < 4; ++j) {
      BF16x16 b;
      const __bf16* wr = Wo + (size_t)(n0 + j * 16 + l16) * ND + kc + bkb;
      b.h[0] = *(const v8bf*)(wr);
      b.h[1] = *(const v8bf*)(wr + 8);
      acc[j] = wmma_bf16(a.v, b.v, acc[j]);
    }
  }

  #pragma unroll
  for (int j = 0; j < 4; ++j) {
    const int col = n0 + j * 16 + l16;
    const float bias = bo[col];
    #pragma unroll
    for (int v = 0; v < 8; ++v) {
      const int m = m0 + v + 8 * half;
      out[(size_t)m * ND + col] = acc[j][v] + bias;
    }
  }
}

// ---------------------------------------------------------------------------
// Launch: cvt -> proj -> attn -> out, all stream-ordered. Workspace ~30 MB.
// Inputs (dict order): content, category, Wqc, Wkc, Wv, Wqk, Wkk, Wo, bo,
// alpha_logit, causal_mask (unused; causality computed analytically).
// ---------------------------------------------------------------------------
extern "C" void kernel_launch(void* const* d_in, const int* in_sizes, int n_in,
                              void* d_out, int out_size, void* d_ws, size_t ws_size,
                              hipStream_t stream) {
  (void)in_sizes; (void)n_in; (void)out_size; (void)ws_size;
  const float* content = (const float*)d_in[0];
  const float* category = (const float*)d_in[1];
  const float* Wmats[6] = {
      (const float*)d_in[2],   // Wqc
      (const float*)d_in[3],   // Wkc
      (const float*)d_in[4],   // Wv
      (const float*)d_in[5],   // Wqk
      (const float*)d_in[6],   // Wkk
      (const float*)d_in[7],   // Wo
  };
  const float* bo = (const float*)d_in[8];
  const float* alpha_logit = (const float*)d_in[9];
  __bf16* ws = (__bf16*)d_ws;
  float* out = (float*)d_out;

  // 0) f32 -> bf16 pre-conversion.
  const int xq4 = (int)(PROJ_ELEMS / 4);       // 524288 quads
  const int wq4 = (int)(WMAT_ELEMS / 4);       // 16384 quads
  dif_cvt_kernel<<<(xq4 + 255) / 256, 256, 0, stream>>>(content,  ws + XC_OFF, xq4);
  dif_cvt_kernel<<<(xq4 + 255) / 256, 256, 0, stream>>>(category, ws + XK_OFF, xq4);
  for (int i = 0; i < 6; ++i) {
    dif_cvt_kernel<<<(wq4 + 255) / 256, 256, 0, stream>>>(
        Wmats[i], ws + WB_OFF + (size_t)i * WMAT_ELEMS, wq4);
  }

  // 1) projections  2) attention  3) output projection
  dif_proj_kernel<<<1280, 256, 0, stream>>>(ws);
  dif_attn_kernel<<<512, 256, 0, stream>>>(alpha_logit, ws);
  dif_out_kernel<<<256, 256, 0, stream>>>(ws, bo, out);
}
